// Attention_11158325035557
// MI455X (gfx1250) — matrix operand order repaired and checked by reading, not compiled, not run on hardware
//
#include <hip/hip_runtime.h>

// ---------------------------------------------------------------------------
// CDNA5 (gfx1250) fused attention:  wave32, V_WMMA_F32_16X16X32_BF16 path.
//   k1: QKV 1x1-conv projections (fp32 -> bf16, WMMA)
//   k2: flash-attention (online softmax) + residual, O(N) memory,
//       K/V tiles staged with GLOBAL_LOAD_ASYNC_TO_LDS (ASYNCcnt-tracked DMA)
// ---------------------------------------------------------------------------

#define USE_ASYNC_LDS 1   // async global->LDS DMA for K/V tiles (inline asm)

typedef __attribute__((ext_vector_type(16))) __bf16        v16bf;
typedef __attribute__((ext_vector_type(8)))  float         v8f;
typedef __attribute__((ext_vector_type(4)))  unsigned int  u32x4;
typedef __attribute__((ext_vector_type(2)))  unsigned int  u32x2;

union Frag16 {                 // 16 bf16 per lane = 8 VGPRs (A or B operand)
  v16bf          v;
  u32x4          q[2];
  unsigned short s[16];
};

#define WMMA_BF16(A, Bm, Cm) \
  __builtin_amdgcn_wmma_f32_16x16x32_bf16(false, (A), false, (Bm), (short)0, (Cm), false, false)

__device__ __forceinline__ unsigned short f2bf(float f) {  // RNE f32 -> bf16
  unsigned int u = __float_as_uint(f);
  u += 0x7FFFu + ((u >> 16) & 1u);
  return (unsigned short)(u >> 16);
}

#if USE_ASYNC_LDS
// Async LDS DMA (CDNA5 §10.7 / 15.18.3): lane supplies LDS byte address (VDST)
// and 64-bit global address (VADDR); completion tracked with ASYNCcnt.
__device__ __forceinline__ void async_ld_b128(const void* gptr, void* lptr) {
  asm volatile("global_load_async_to_lds_b128 %0, %1, off"
               :
               : "v"((unsigned int)(unsigned long long)lptr),
                 "v"((unsigned long long)gptr)
               : "memory");
}
__device__ __forceinline__ void async_ld_b64(const void* gptr, void* lptr) {
  asm volatile("global_load_async_to_lds_b64 %0, %1, off"
               :
               : "v"((unsigned int)(unsigned long long)lptr),
                 "v"((unsigned long long)gptr)
               : "memory");
}
__device__ __forceinline__ void wait_async0() {
  asm volatile("s_wait_asynccnt 0x0" ::: "memory");
}
#endif

static constexpr int CC = 256;   // channels
static constexpr int DD = 32;    // q/k head dim
static constexpr int NN = 4096;  // spatial
static constexpr int BB = 4;     // batch
static constexpr int XP = 264;   // padded LDS row (ushorts): bank-stride 4

// ============================================================================
// Kernel 1: Q = x^T w1^T + b1  [B,N,32]  (bf16)
//           K = x^T w2^T + b2  [B,N,32]  (bf16)   (n-major, flash reads rows)
//           V = w3 x + b3      [B,C,N]   (bf16)   (c-major, PV B-frag contiguous)
// ============================================================================
__global__ __launch_bounds__(256)
void qkv_proj_kernel(const float* __restrict__ x,
                     const float* __restrict__ w1, const float* __restrict__ b1,
                     const float* __restrict__ w2, const float* __restrict__ b2,
                     const float* __restrict__ w3, const float* __restrict__ b3,
                     unsigned short* __restrict__ qws,
                     unsigned short* __restrict__ kws,
                     unsigned short* __restrict__ vws)
{
  __shared__ unsigned short xT[64 * XP];  // x^T tile [64 n][256 c] bf16, padded

  const int tid  = threadIdx.x;
  const int b    = blockIdx.x >> 6;
  const int n0   = (blockIdx.x & 63) * 64;
  const int w    = tid >> 5;
  const int lane = tid & 31;
  const int half = lane >> 4;
  const int ln   = lane & 15;

  // ---- stage x^T tile (coalesced global read, bank-friendly LDS write) ----
  for (int i = tid; i < 256 * 64; i += 256) {
    int c = i >> 6, nl = i & 63;
    xT[nl * XP + c] = f2bf(x[(((b << 8) + c) << 12) + n0 + nl]);
  }
  __syncthreads();

  // ---- V: 16 c-tiles x 4 n-tiles = 64 output tiles, K=256 (8 WMMA each) ----
  for (int t = w; t < 64; t += 8) {
    int ct = t >> 2, nt = t & 3;
    v8f acc = {};
#pragma unroll
    for (int kk = 0; kk < 8; ++kk) {
      int ck = kk * 32;
      Frag16 a;  // A = w3 tile [16 c x 32 c']: row = lane&15
      const float* wr = w3 + ((ct * 16 + ln) << 8) + ck + half * 16;
#pragma unroll
      for (int j = 0; j < 8; ++j) { a.s[j] = f2bf(wr[j]); a.s[8 + j] = f2bf(wr[8 + j]); }
      Frag16 bm; // B = x tile [32 c' x 16 n]: col = lane&15, K = half*16+t
      const unsigned short* xr = xT + (nt * 16 + ln) * XP + ck + half * 16;
      bm.q[0] = *(const u32x4*)(xr);
      bm.q[1] = *(const u32x4*)(xr + 8);
      acc = WMMA_BF16(a.v, bm.v, acc);
    }
#pragma unroll
    for (int r = 0; r < 8; ++r) {                 // D: row c = ct*16+r+8*half
      int c = ct * 16 + r + 8 * half;
      vws[(((b << 8) + c) << 12) + n0 + nt * 16 + ln] = f2bf(acc[r] + b3[c]);
    }
  }

  // ---- Q and K: 4 n-tiles x 2 d-tiles; one tile pair per wave ----
  {
    int nt = w & 3, dt = w >> 2;
    v8f aq = {}, ak = {};
#pragma unroll
    for (int kk = 0; kk < 8; ++kk) {
      int ck = kk * 32;
      Frag16 a;  // A = x^T tile [16 n x 32 c']
      const unsigned short* xr = xT + (nt * 16 + ln) * XP + ck;
      a.q[0] = *(const u32x4*)(xr + half * 16);
      a.q[1] = *(const u32x4*)(xr + 8 + half * 16);
      Frag16 f1, f2; // B = w^T tile [32 c' x 16 d]: lane reads contiguous w-row
      const float* r1 = w1 + ((dt * 16 + ln) << 8) + ck + half * 16;
      const float* r2 = w2 + ((dt * 16 + ln) << 8) + ck + half * 16;
#pragma unroll
      for (int tt = 0; tt < 16; ++tt) { f1.s[tt] = f2bf(r1[tt]); f2.s[tt] = f2bf(r2[tt]); }
      aq = WMMA_BF16(a.v, f1.v, aq);
      ak = WMMA_BF16(a.v, f2.v, ak);
    }
#pragma unroll
    for (int r = 0; r < 8; ++r) {
      int n = n0 + nt * 16 + r + 8 * half;
      int d = dt * 16 + ln;
      qws[(((b << 12) + n) << 5) + d] = f2bf(aq[r] + b1[d]);
      kws[(((b << 12) + n) << 5) + d] = f2bf(ak[r] + b2[d]);
    }
  }
}

// ============================================================================
// Kernel 2: flash attention. Block = 8 waves = 128 query rows; 128 key steps
// of 32 keys. Per wave per step: 2 WMMA (S) + 16 WMMA (PV).
// K/V tiles double-buffered in LDS via async DMA (ASYNCcnt).
// ============================================================================
static constexpr int    KT_ROW  = 40;                    // padded row (ushorts)
static constexpr int    KT_BUF  = 32 * KT_ROW;           // K tile buffer
static constexpr int    VT_OFF  = 2 * KT_BUF;
static constexpr int    VT_BUF  = 256 * KT_ROW;          // V tile buffer
static constexpr int    PT_OFF  = VT_OFF + 2 * VT_BUF;   // per-wave P scratch
static constexpr int    PT_W    = 16 * KT_ROW;
static constexpr int    ONP     = 132;                   // O staging row pad
static constexpr size_t SMEM_BYTES = (size_t)256 * ONP * 4;  // 132 KB (covers all)

__global__ __launch_bounds__(256)
void flash_attn_kernel(const float* __restrict__ x,
                       const unsigned short* __restrict__ qws,
                       const unsigned short* __restrict__ kws,
                       const unsigned short* __restrict__ vws,
                       float* __restrict__ out)
{
  extern __shared__ unsigned short smem[];

  const int tid  = threadIdx.x;
  const int b    = blockIdx.x >> 5;
  const int n0   = (blockIdx.x & 31) << 7;   // 128 query rows per block
  const int w    = tid >> 5;
  const int lane = tid & 31;
  const int half = lane >> 4;
  const int ln   = lane & 15;
  const int qn0  = n0 + w * 16;

  // Q fragment [16 n x 32 d], resident all 128 iterations
  Frag16 qf;
  {
    const unsigned short* qr = qws + (((b << 12) + qn0 + ln) << 5);
    qf.q[0] = *(const u32x4*)(qr + half * 16);
    qf.q[1] = *(const u32x4*)(qr + 8 + half * 16);
  }

  v8f acc[16];                              // O accumulator: 16 n x 256 c fp32
  v8f zero = {};
#pragma unroll
  for (int i = 0; i < 16; ++i) acc[i] = zero;
  float mrow[8], lrow[8];
#pragma unroll
  for (int r = 0; r < 8; ++r) { mrow[r] = -3.0e38f; lrow[r] = 0.0f; }

  const unsigned short* kbase = kws + (((size_t)b) << 17);        // b*4096*32
  const unsigned short* vbase = vws + (((size_t)b) << 20);        // b*256*4096

#if USE_ASYNC_LDS
  // Issue this thread's share of the K and V tile copies for (buf, it):
  //   K tile: 32 rows x 32 d  -> thread moves 8 B  (b64)
  //   V tile: 256 c  x 32 m   -> thread moves 64 B (4 x b128), row c = tid
  auto issue_tiles = [&](int buf, int it) {
    int m0 = it << 5;
    const unsigned short* kg = kbase + (((m0 + (tid >> 3)) << 5) + ((tid & 7) << 2));
    unsigned short* kd = smem + buf * KT_BUF + (tid >> 3) * KT_ROW + ((tid & 7) << 2);
    async_ld_b64(kg, kd);
    const unsigned short* vg = vbase + (((size_t)tid) << 12) + m0;
    unsigned short* vd = smem + VT_OFF + buf * VT_BUF + tid * KT_ROW;
#pragma unroll
    for (int j = 0; j < 4; ++j) async_ld_b128(vg + 8 * j, vd + 8 * j);
  };

  issue_tiles(0, 0);
  wait_async0();
  __syncthreads();
#else
  // Fallback: software-pipelined global->reg->LDS double buffering
  u32x2 kreg;
  u32x4 vreg0, vreg1, vreg2, vreg3;
  auto ldg = [&](int it) {
    int m0 = it << 5;
    kreg = *(const u32x2*)(kbase + (((m0 + (tid >> 3)) << 5) + ((tid & 7) << 2)));
    const unsigned short* vr = vbase + (((size_t)tid) << 12) + m0;
    vreg0 = ((const u32x4*)vr)[0];
    vreg1 = ((const u32x4*)vr)[1];
    vreg2 = ((const u32x4*)vr)[2];
    vreg3 = ((const u32x4*)vr)[3];
  };
  auto sts = [&](int buf) {
    unsigned short* kd = smem + buf * KT_BUF + (tid >> 3) * KT_ROW + ((tid & 7) << 2);
    *(u32x2*)kd = kreg;
    unsigned short* vd = smem + VT_OFF + buf * VT_BUF + tid * KT_ROW;
    *(u32x4*)(vd + 0)  = vreg0;
    *(u32x4*)(vd + 8)  = vreg1;
    *(u32x4*)(vd + 16) = vreg2;
    *(u32x4*)(vd + 24) = vreg3;
  };
  ldg(0);
  sts(0);
  __syncthreads();
#endif

  for (int it = 0; it < 128; ++it) {
    const int buf = it & 1;
#if USE_ASYNC_LDS
    if (it + 1 < 128) issue_tiles(buf ^ 1, it + 1);  // DMA overlaps WMMA work
#else
    if (it + 1 < 128) ldg(it + 1);
#endif

    const unsigned short* kt = smem + buf * KT_BUF;
    const unsigned short* vt = smem + VT_OFF + buf * VT_BUF;
    unsigned short*       pt = smem + PT_OFF + w * PT_W;

    // ---- S = Q K^T : two 16-key tiles, one WMMA each (K-dim = d = 32) ----
    Frag16 kf0, kf1;  // B-frag: col = key (lane&15), K = d = half*16 + t
    {
      const unsigned short* r0 = kt + ln * KT_ROW + half * 16;
      kf0.q[0] = *(const u32x4*)(r0);
      kf0.q[1] = *(const u32x4*)(r0 + 8);
      const unsigned short* r1 = kt + (16 + ln) * KT_ROW + half * 16;
      kf1.q[0] = *(const u32x4*)(r1);
      kf1.q[1] = *(const u32x4*)(r1 + 8);
    }
    v8f s0 = WMMA_BF16(qf.v, kf0.v, zero);
    v8f s1 = WMMA_BF16(qf.v, kf1.v, zero);

    // ---- online softmax over this 32-key slab (rows live across 16 lanes) --
    float corr[8];
#pragma unroll
    for (int r = 0; r < 8; ++r) {
      float mx = fmaxf(s0[r], s1[r]);
      mx = fmaxf(mx, __shfl_xor(mx, 1, 32));
      mx = fmaxf(mx, __shfl_xor(mx, 2, 32));
      mx = fmaxf(mx, __shfl_xor(mx, 4, 32));
      mx = fmaxf(mx, __shfl_xor(mx, 8, 32));
      float mnew = fmaxf(mrow[r], mx);
      float cr   = __expf(mrow[r] - mnew);
      mrow[r]    = mnew;
      float p0 = __expf(s0[r] - mnew);
      float p1 = __expf(s1[r] - mnew);
      float rs = p0 + p1;
      rs += __shfl_xor(rs, 1, 32);
      rs += __shfl_xor(rs, 2, 32);
      rs += __shfl_xor(rs, 4, 32);
      rs += __shfl_xor(rs, 8, 32);
      lrow[r] = lrow[r] * cr + rs;
      corr[r] = cr;
      int prow = (r + 8 * half) * KT_ROW;   // transpose P via per-wave LDS
      pt[prow + ln]      = f2bf(p0);
      pt[prow + 16 + ln] = f2bf(p1);
    }
    asm volatile("s_wait_dscnt 0" ::: "memory");  // P stores visible to reads

    Frag16 pf;  // A-frag of P [16 n x 32 keys]
    {
      const unsigned short* pr = pt + ln * KT_ROW;
      pf.q[0] = *(const u32x4*)(pr + half * 16);
      pf.q[1] = *(const u32x4*)(pr + 8 + half * 16);
    }

    // ---- O = diag(corr) O + P V : 16 WMMAs (one per 16-channel tile) ----
#pragma unroll
    for (int ct = 0; ct < 16; ++ct) {
      v8f a = acc[ct];
#pragma unroll
      for (int r = 0; r < 8; ++r) a[r] *= corr[r];
      Frag16 vf;  // B-frag: col = channel (lane&15), K = key = half*16 + t
      const unsigned short* vr = vt + (ct * 16 + ln) * KT_ROW + half * 16;
      vf.q[0] = *(const u32x4*)(vr);
      vf.q[1] = *(const u32x4*)(vr + 8);
      acc[ct] = WMMA_BF16(pf.v, vf.v, a);
    }

#if USE_ASYNC_LDS
    if (it + 1 < 128) wait_async0();   // next tiles resident before barrier
#else
    if (it + 1 < 128) sts(buf ^ 1);
#endif
    __syncthreads();
  }

  // ---- epilogue: normalize, transpose through LDS, coalesced store + resid --
  __syncthreads();
  float* oblk = (float*)smem;               // reuse pool: [256 c][128 n] padded
#pragma unroll
  for (int r = 0; r < 8; ++r) lrow[r] = __frcp_rn(lrow[r]);
#pragma unroll
  for (int ct = 0; ct < 16; ++ct) {
#pragma unroll
    for (int r = 0; r < 8; ++r) {
      int c  = ct * 16 + ln;
      int nl = w * 16 + r + 8 * half;
      oblk[c * ONP + nl] = acc[ct][r] * lrow[r];
    }
  }
  __syncthreads();
  for (int idx = tid; idx < 256 * 128; idx += 256) {
    int c = idx >> 7, nl = idx & 127;
    int g = (((b << 8) + c) << 12) + n0 + nl;
    out[g] = oblk[c * ONP + nl] + x[g];
  }
}

// ============================================================================
extern "C" void kernel_launch(void* const* d_in, const int* in_sizes, int n_in,
                              void* d_out, int out_size, void* d_ws, size_t ws_size,
                              hipStream_t stream) {
  (void)in_sizes; (void)n_in; (void)out_size; (void)ws_size;
  const float* x  = (const float*)d_in[0];
  const float* w1 = (const float*)d_in[1];
  const float* b1 = (const float*)d_in[2];
  const float* w2 = (const float*)d_in[3];
  const float* b2 = (const float*)d_in[4];
  const float* w3 = (const float*)d_in[5];
  const float* b3 = (const float*)d_in[6];

  // workspace: Q [4,4096,32] bf16 | K [4,4096,32] bf16 | V [4,256,4096] bf16
  unsigned short* qws = (unsigned short*)d_ws;
  unsigned short* kws = qws + (size_t)BB * NN * DD;
  unsigned short* vws = kws + (size_t)BB * NN * DD;

  qkv_proj_kernel<<<dim3(BB * (NN / 64)), dim3(256), 0, stream>>>(
      x, w1, b1, w2, b2, w3, b3, qws, kws, vws);

  flash_attn_kernel<<<dim3(BB * (NN / 128)), dim3(256), SMEM_BYTES, stream>>>(
      x, qws, kws, vws, (float*)d_out);
}
